// Model_53523882443412
// MI455X (gfx1250) — compile-verified, matrix operand order checked by reading, not run on hardware
//
#include <hip/hip_runtime.h>
#include <math.h>

// ---------------------------------------------------------------------------
// GraphSAGE actor/critic forward for MI455X (gfx1250, wave32, WMMA).
// Dense layers run as fused K=256 GEMMs on v_wmma_f32_16x16x32_bf16.
// Edge aggregation = f32 global atomics (L2-resident: 25.6MB node table << 192MB L2).
// ---------------------------------------------------------------------------

typedef __attribute__((ext_vector_type(16))) __bf16 v16bf;
typedef __attribute__((ext_vector_type(8)))  __bf16 v8bf;
typedef __attribute__((ext_vector_type(8)))  float  v8f;

#define UNITS 128
#define IN_F  5
#define RED_BLOCKS 200

// ---------------- degree / inv_deg ----------------
__global__ void degree_kernel(const int* __restrict__ dst, float* __restrict__ deg, int E) {
    int e = blockIdx.x * blockDim.x + threadIdx.x;
    if (e < E) atomicAdd(&deg[dst[e]], 1.0f);
}

__global__ void invdeg_kernel(const float* __restrict__ deg, float* __restrict__ invdeg, int n) {
    int i = blockIdx.x * blockDim.x + threadIdx.x;
    if (i < n) invdeg[i] = 1.0f / fmaxf(deg[i], 1.0f);
}

// ---------------- layer-1 (5-wide) aggregation + dense ----------------
__global__ void agg5_kernel(const float* __restrict__ x, const int* __restrict__ src,
                            const int* __restrict__ dst, float* __restrict__ aggx, int E) {
    int e = blockIdx.x * blockDim.x + threadIdx.x;
    if (e >= E) return;
    int s = src[e], d = dst[e];
    #pragma unroll
    for (int f = 0; f < IN_F; ++f)
        atomicAdd(&aggx[(size_t)d * IN_F + f], x[(size_t)s * IN_F + f]);
}

__global__ void layer1_kernel(const float* __restrict__ x, const float* __restrict__ aggx,
                              const float* __restrict__ invdeg,
                              const float* __restrict__ Wl, const float* __restrict__ bl,
                              const float* __restrict__ Wr, float* __restrict__ h1, int n) {
    int t = blockIdx.x * blockDim.x + threadIdx.x;
    if (t >= n * UNITS) return;
    int i = t >> 7, c = t & (UNITS - 1);
    float s = invdeg[i];
    float acc = bl[c];
    #pragma unroll
    for (int k = 0; k < IN_F; ++k) {
        float a = aggx[(size_t)i * IN_F + k] * s;
        float v = x[(size_t)i * IN_F + k];
        acc += a * Wl[k * UNITS + c] + v * Wr[k * UNITS + c];
    }
    h1[t] = tanhf(acc);
}

// ---------------- 128-wide edge aggregation: wave per edge, 4 floats/lane --
__global__ void agg128_kernel(const float* __restrict__ h, const int* __restrict__ src,
                              const int* __restrict__ dst, float* __restrict__ agg, int E) {
    int t = blockIdx.x * blockDim.x + threadIdx.x;
    int e = t >> 5;           // 32 lanes per edge
    int lane = t & 31;
    if (e >= E) return;
    int s = src[e], d = dst[e];
    const float4 v = *(const float4*)(h + (size_t)s * UNITS + lane * 4);
    float* ad = agg + (size_t)d * UNITS + lane * 4;
    atomicAdd(ad + 0, v.x);
    atomicAdd(ad + 1, v.y);
    atomicAdd(ad + 2, v.z);
    atomicAdd(ad + 3, v.w);
}

// ---------------- build fused A = [agg*invdeg | h] in bf16, [n,256] -------
__global__ void build_A_kernel(const float* __restrict__ aggH, const float* __restrict__ invdeg,
                               const float* __restrict__ h, __bf16* __restrict__ A, int n) {
    int t = blockIdx.x * blockDim.x + threadIdx.x;
    if (t >= n * UNITS) return;
    int i = t >> 7, c = t & (UNITS - 1);
    float s = invdeg[i];
    A[(size_t)i * 256 + c]         = (__bf16)(aggH[t] * s);
    A[(size_t)i * 256 + UNITS + c] = (__bf16)h[t];
}

// ---------------- pack [Wl;Wr] (each 128x128 f32) into WMMA B-fragment order
// layout: Bp[((coltile*8 + kstep)*32 + lane)*16 + j], bf16
// per ISA 16-bit B layout: lane<16 -> kbase=0, lane>=16 -> kbase=8; j>=8 -> +16
__global__ void packW_kernel(const float* __restrict__ Wl, const float* __restrict__ Wr,
                             __bf16* __restrict__ Bp) {
    int t = blockIdx.x * blockDim.x + threadIdx.x;   // 8*8*32*16 = 32768
    if (t >= 256 * UNITS) return;
    int j  = t & 15;
    int L  = (t >> 4) & 31;
    int kk = (t >> 9) & 7;
    int c  = t >> 12;
    int k  = kk * 32 + ((L < 16) ? 0 : 8) + j + ((j >= 8) ? 8 : 0);   // 0..255
    int col = c * 16 + (L & 15);
    float v = (k < UNITS) ? Wl[k * UNITS + col] : Wr[(k - UNITS) * UNITS + col];
    Bp[t] = (__bf16)v;
}

// ---------------- WMMA GEMM: out[n,128] = A[n,256] @ Wcat[256,128] + bias --
// block = 256 threads = 8 waves; each wave owns one 16-col tile; A tile in LDS.
__global__ void __launch_bounds__(256)
gemm_wmma_kernel(const __bf16* __restrict__ A, const __bf16* __restrict__ Bp,
                 const float* __restrict__ bias, float* __restrict__ out,
                 int n, int do_tanh) {
    __shared__ __bf16 ldsA[16 * 256];     // 8 KB A tile
    const int tid  = threadIdx.x;
    const int wave = tid >> 5;
    const int lane = tid & 31;
    const int row0 = blockIdx.x * 16;     // n is a multiple of 16 here (50000)

    // cooperative A-tile load: 8192 B = 512 x uint4, 2 per thread
    {
        const uint4* src = (const uint4*)(A + (size_t)row0 * 256);
        uint4* dst = (uint4*)ldsA;
        dst[tid]       = src[tid];
        dst[tid + 256] = src[tid + 256];
    }
    __syncthreads();

    const int m  = lane & 15;             // A row within tile
    const int kb = (lane < 16) ? 0 : 8;   // per-lane K base (16-bit A layout)
    v8f c = {};
    const __bf16* bptr = Bp + ((size_t)(wave * 8) * 32 + lane) * 16;

    #pragma unroll
    for (int kk = 0; kk < 8; ++kk) {      // K = 256 in 8 steps of 32
        v8bf alo = *(const v8bf*)(&ldsA[m * 256 + kk * 32 + kb]);        // K kb..kb+7
        v8bf ahi = *(const v8bf*)(&ldsA[m * 256 + kk * 32 + kb + 16]);   // K kb+16..kb+23
        v16bf a = __builtin_shufflevector(alo, ahi,
                    0,1,2,3,4,5,6,7,8,9,10,11,12,13,14,15);
        v16bf b = *(const v16bf*)(bptr + (size_t)kk * 32 * 16);          // contiguous 32 B
        c = __builtin_amdgcn_wmma_f32_16x16x32_bf16(
                false, a, false, b, (short)0, c, false, false);
    }

    // C/D layout: lanes 0-15 rows 0..7, lanes 16-31 rows 8..15; col = lane&15
    const int col = wave * 16 + (lane & 15);
    const int mo  = (lane < 16) ? 0 : 8;
    const float bv = bias[col];
    #pragma unroll
    for (int r = 0; r < 8; ++r) {
        int row = row0 + r + mo;
        if (row < n) {
            float v = c[r] + bv;
            if (do_tanh) v = tanhf(v);
            out[(size_t)row * UNITS + col] = v;
        }
    }
}

// ---------------- actor head: scalar dot (N=1 output, WMMA is wasteful) ---
__global__ void actor_kernel(const float* __restrict__ aggH, const float* __restrict__ invdeg,
                             const float* __restrict__ h2, const float* __restrict__ Wla,
                             const float* __restrict__ bla, const float* __restrict__ Wra,
                             const float* __restrict__ x, float* __restrict__ actor_raw, int n) {
    int i = blockIdx.x * blockDim.x + threadIdx.x;
    if (i >= n) return;
    float s = invdeg[i];
    float acc = bla[0];
    const float* ar = aggH + (size_t)i * UNITS;
    const float* hr = h2   + (size_t)i * UNITS;
    #pragma unroll 8
    for (int k = 0; k < UNITS; ++k)
        acc += ar[k] * s * Wla[k] + hr[k] * Wra[k];
    if (x[(size_t)i * IN_F + 2] != 0.0f) acc = -INFINITY;   // do_not_flip mask
    actor_raw[i] = acc;
}

// ---------------- critic reduce: dot with Wf, atomic sum, tanh(mean) ------
__global__ void critic_dot_kernel(const float* __restrict__ crit128, const float* __restrict__ Wf,
                                  const float* __restrict__ bfv, float* __restrict__ csum, int n) {
    int i = blockIdx.x * blockDim.x + threadIdx.x;
    if (i >= n) return;
    float acc = bfv[0];
    const float* cr = crit128 + (size_t)i * UNITS;
    #pragma unroll 8
    for (int k = 0; k < UNITS; ++k) acc += cr[k] * Wf[k];
    atomicAdd(csum, acc);
}

__global__ void critic_final_kernel(const float* __restrict__ csum, float* __restrict__ out, int n) {
    if (blockIdx.x == 0 && threadIdx.x == 0)
        out[n] = tanhf(csum[0] / fmaxf((float)n, 1.0f));
}

// ---------------- log_softmax over all n nodes (2-pass reduction) ---------
__global__ void reduce_max_kernel(const float* __restrict__ v, float* __restrict__ partial, int n) {
    __shared__ float sm[256];
    float m = -INFINITY;
    for (int i = blockIdx.x * 256 + threadIdx.x; i < n; i += gridDim.x * 256)
        m = fmaxf(m, v[i]);
    sm[threadIdx.x] = m; __syncthreads();
    for (int s = 128; s > 0; s >>= 1) {
        if (threadIdx.x < s) sm[threadIdx.x] = fmaxf(sm[threadIdx.x], sm[threadIdx.x + s]);
        __syncthreads();
    }
    if (threadIdx.x == 0) partial[blockIdx.x] = sm[0];
}

__global__ void final_max_kernel(const float* __restrict__ partial, int nb, float* __restrict__ gmax) {
    __shared__ float sm[256];
    float m = (threadIdx.x < nb) ? partial[threadIdx.x] : -INFINITY;
    sm[threadIdx.x] = m; __syncthreads();
    for (int s = 128; s > 0; s >>= 1) {
        if (threadIdx.x < s) sm[threadIdx.x] = fmaxf(sm[threadIdx.x], sm[threadIdx.x + s]);
        __syncthreads();
    }
    if (threadIdx.x == 0) gmax[0] = sm[0];
}

__global__ void reduce_sumexp_kernel(const float* __restrict__ v, const float* __restrict__ gmax,
                                     float* __restrict__ partial, int n) {
    __shared__ float sm[256];
    float gm = gmax[0];
    float acc = 0.0f;
    for (int i = blockIdx.x * 256 + threadIdx.x; i < n; i += gridDim.x * 256)
        acc += __expf(v[i] - gm);       // exp(-inf - gm) = 0 handles masked rows
    sm[threadIdx.x] = acc; __syncthreads();
    for (int s = 128; s > 0; s >>= 1) {
        if (threadIdx.x < s) sm[threadIdx.x] += sm[threadIdx.x + s];
        __syncthreads();
    }
    if (threadIdx.x == 0) partial[blockIdx.x] = sm[0];
}

__global__ void final_logZ_kernel(const float* __restrict__ partial, int nb,
                                  const float* __restrict__ gmax, float* __restrict__ logZ) {
    __shared__ float sm[256];
    sm[threadIdx.x] = (threadIdx.x < nb) ? partial[threadIdx.x] : 0.0f;
    __syncthreads();
    for (int s = 128; s > 0; s >>= 1) {
        if (threadIdx.x < s) sm[threadIdx.x] += sm[threadIdx.x + s];
        __syncthreads();
    }
    if (threadIdx.x == 0) logZ[0] = gmax[0] + logf(sm[0]);
}

__global__ void write_actor_kernel(const float* __restrict__ v, const float* __restrict__ logZ,
                                   float* __restrict__ out, int n) {
    int i = blockIdx.x * blockDim.x + threadIdx.x;
    if (i < n) out[i] = v[i] - logZ[0];
}

// ---------------------------------------------------------------------------
static inline size_t alignUp(size_t x, size_t a) { return (x + a - 1) & ~(a - 1); }

extern "C" void kernel_launch(void* const* d_in, const int* in_sizes, int n_in,
                              void* d_out, int out_size, void* d_ws, size_t ws_size,
                              hipStream_t stream) {
    const float* x   = (const float*)d_in[0];
    const int*   ei  = (const int*)d_in[1];
    const float* Wl1 = (const float*)d_in[3];
    const float* bl1 = (const float*)d_in[4];
    const float* Wr1 = (const float*)d_in[5];
    const float* Wl2 = (const float*)d_in[6];
    const float* bl2 = (const float*)d_in[7];
    const float* Wr2 = (const float*)d_in[8];
    const float* Wla = (const float*)d_in[9];
    const float* bla = (const float*)d_in[10];
    const float* Wra = (const float*)d_in[11];
    const float* Wlc = (const float*)d_in[12];
    const float* blc = (const float*)d_in[13];
    const float* Wrc = (const float*)d_in[14];
    const float* Wf  = (const float*)d_in[15];
    const float* bfv = (const float*)d_in[16];

    const int n = in_sizes[0] / IN_F;     // 50000
    const int E = in_sizes[1] / 2;        // 800000
    const int* srcI = ei;
    const int* dstI = ei + E;
    float* out = (float*)d_out;           // [0..n) actor log-softmax, [n] critic

    // workspace layout
    char* ws = (char*)d_ws;
    size_t off = 0;
    auto take = [&](size_t bytes) { size_t o = off; off = alignUp(off + bytes, 256); return (void*)(ws + o); };
    float*  deg     = (float*)take((size_t)n * 4);
    float*  invdeg  = (float*)take((size_t)n * 4);
    float*  aggx    = (float*)take((size_t)n * IN_F * 4);
    float*  h1      = (float*)take((size_t)n * UNITS * 4);
    float*  aggH    = (float*)take((size_t)n * UNITS * 4);
    float*  h2      = (float*)take((size_t)n * UNITS * 4);
    __bf16* Abf     = (__bf16*)take((size_t)n * 256 * 2);
    __bf16* Bp      = (__bf16*)take((size_t)256 * UNITS * 2);
    float*  actorv  = (float*)take((size_t)n * 4);
    float*  partial = (float*)take((size_t)RED_BLOCKS * 4);
    float*  scal    = (float*)take(16 * 4);     // [0]=gmax [1]=logZ [2]=csum
    float*  crit128 = h1;                       // h1 dead after layer-2 GEMM; alias

    const int B = 256;
    const int gN    = (n + B - 1) / B;
    const int gE    = (E + B - 1) / B;
    const int gNF   = (n * UNITS + B - 1) / B;
    const int gE32  = ((long long)E * 32 + B - 1) / B;
    const int gGemm = (n + 15) / 16;

    // degree + inv_deg
    hipMemsetAsync(deg, 0, (size_t)n * 4, stream);
    degree_kernel<<<gE, B, 0, stream>>>(dstI, deg, E);
    invdeg_kernel<<<gN, B, 0, stream>>>(deg, invdeg, n);

    // layer 1
    hipMemsetAsync(aggx, 0, (size_t)n * IN_F * 4, stream);
    agg5_kernel<<<gE, B, 0, stream>>>(x, srcI, dstI, aggx, E);
    layer1_kernel<<<gNF, B, 0, stream>>>(x, aggx, invdeg, Wl1, bl1, Wr1, h1, n);

    // layer 2 (WMMA)
    hipMemsetAsync(aggH, 0, (size_t)n * UNITS * 4, stream);
    agg128_kernel<<<gE32, B, 0, stream>>>(h1, srcI, dstI, aggH, E);
    build_A_kernel<<<gNF, B, 0, stream>>>(aggH, invdeg, h1, Abf, n);
    packW_kernel<<<(256 * UNITS + B - 1) / B, B, 0, stream>>>(Wl2, Wr2, Bp);
    gemm_wmma_kernel<<<gGemm, B, 0, stream>>>(Abf, Bp, bl2, h2, n, /*tanh=*/1);

    // shared aggregate of h2 for actor + critic heads
    hipMemsetAsync(aggH, 0, (size_t)n * UNITS * 4, stream);
    agg128_kernel<<<gE32, B, 0, stream>>>(h2, srcI, dstI, aggH, E);

    // critic 128-wide layer (WMMA), then dot + mean + tanh
    build_A_kernel<<<gNF, B, 0, stream>>>(aggH, invdeg, h2, Abf, n);
    packW_kernel<<<(256 * UNITS + B - 1) / B, B, 0, stream>>>(Wlc, Wrc, Bp);
    gemm_wmma_kernel<<<gGemm, B, 0, stream>>>(Abf, Bp, blc, crit128, n, /*tanh=*/0);
    hipMemsetAsync(scal + 2, 0, 4, stream);
    critic_dot_kernel<<<gN, B, 0, stream>>>(crit128, Wf, bfv, scal + 2, n);
    critic_final_kernel<<<1, 64, 0, stream>>>(scal + 2, out, n);

    // actor head + masked log_softmax over all nodes
    actor_kernel<<<gN, B, 0, stream>>>(aggH, invdeg, h2, Wla, bla, Wra, x, actorv, n);
    reduce_max_kernel<<<RED_BLOCKS, B, 0, stream>>>(actorv, partial, n);
    final_max_kernel<<<1, B, 0, stream>>>(partial, RED_BLOCKS, scal + 0);
    reduce_sumexp_kernel<<<RED_BLOCKS, B, 0, stream>>>(actorv, scal + 0, partial, n);
    final_logZ_kernel<<<1, B, 0, stream>>>(partial, RED_BLOCKS, scal + 0, scal + 1);
    write_actor_kernel<<<gN, B, 0, stream>>>(actorv, scal + 1, out, n);

    (void)n_in; (void)out_size; (void)ws_size;
}
